// STFlayer_11733850652885
// MI455X (gfx1250) — matrix operand order repaired and checked by reading, not compiled, not run on hardware
//
#include <hip/hip_runtime.h>
#include <hip/hip_bf16.h>

typedef __attribute__((ext_vector_type(16))) _Float16 v16h;
typedef __attribute__((ext_vector_type(8)))  float    v8f;
typedef int v4i __attribute__((vector_size(16)));
typedef __attribute__((address_space(3))) v4i lds_v4i;

namespace cfg {
constexpr int BATCH = 64;
constexpr int CIN   = 16;
constexpr int LEN   = 8192;
constexpr int N0    = 64,  N1 = 256;          // FFT sizes
constexpr int F0    = 33,  F1 = 129;          // rfft bins
constexpr int T0    = 128, T1 = 32;           // kept frames
constexpr int CO    = 32;
constexpr int RATIO = 4;
constexpr float EPS = 1e-5f;

constexpr int R0  = BATCH * CIN * T0;         // 131072 stft rows (fft0)
constexpr int R1  = BATCH * CIN * T1;         // 32768  stft rows (fft1)
constexpr int NP0 = 80;                        // pad(2*F0=66 -> mult 16)
constexpr int NP1 = 272;                       // pad(2*F1=258 -> mult 16)
constexpr int KI0 = 96;                        // pad(66 -> mult 32)   istft K
constexpr int KI1 = 288;                       // pad(258 -> mult 32)  istft K
constexpr int EM0 = BATCH * T0;                // 8192 einsum rows
constexpr int EM1 = BATCH * T1;                // 2048 einsum rows
constexpr int WPB = 8;                         // waves per block for WMMA kernels
constexpr float PI2 = 6.283185307179586f;
}
using namespace cfg;

// ---------------- WMMA fragment loaders (CDNA5 wave32 layouts) ----------------
// A fragment 16x32 f16: lane L -> M=L&15 ; VGPR r -> K = (r&4?16:0)+(L&16?8:0)+(r&3)*2, +1
__device__ __forceinline__ v16h load_a_frag(const _Float16* lane_row, int kbase, int lane) {
  v8f bits;
  const int khalf = (lane & 16) ? 8 : 0;
#pragma unroll
  for (int r = 0; r < 8; ++r) {
    int k = kbase + ((r & 4) ? 16 : 0) + khalf + ((r & 3) << 1);
    bits[r] = *(const float*)(lane_row + k);
  }
  return __builtin_bit_cast(v16h, bits);
}

// B fragment 32x16 f16 stored transposed [N][Kpad]:
// lane L -> N=L&15 ; VGPR r -> K = (r&4?16:0)+(L&16?8:0)+(r&3)*2, +1
__device__ __forceinline__ v16h load_b_frag(const _Float16* nt_base, int kpad, int kbase, int lane) {
  v8f bits;
  const _Float16* p = nt_base + (long)(lane & 15) * kpad + kbase + ((lane & 16) ? 8 : 0);
#pragma unroll
  for (int r = 0; r < 8; ++r) bits[r] = *(const float*)(p + ((r & 4) ? 16 : 0) + ((r & 3) << 1));
  return __builtin_bit_cast(v16h, bits);
}

// Cooperative stage of one 16-row B tile (16*kp halfs, contiguous in global) into LDS.
// Uses CDNA5 async global->LDS DMA when the builtin is available.
__device__ __forceinline__ void stage_b_tile(const _Float16* gsrc, _Float16* lds, int halfs) {
  int chunks = halfs >> 3;  // 16-byte chunks
#if __has_builtin(__builtin_amdgcn_global_load_async_to_lds_b128)
  for (int i = threadIdx.x; i < chunks; i += blockDim.x) {
    __builtin_amdgcn_global_load_async_to_lds_b128(
        (v4i*)(gsrc + (i << 3)),
        (lds_v4i*)(lds + (i << 3)), 0, 0);
  }
#if __has_builtin(__builtin_amdgcn_s_wait_asynccnt)
  __builtin_amdgcn_s_wait_asynccnt(0);
#else
  asm volatile("s_wait_asynccnt 0" ::: "memory");
#endif
#else
  for (int i = threadIdx.x; i < chunks; i += blockDim.x)
    ((float4*)lds)[i] = ((const float4*)gsrc)[i];
#endif
#if __has_builtin(__builtin_amdgcn_s_wait_tensorcnt)
  __builtin_amdgcn_s_wait_tensorcnt(0);  // TENSORcnt==0 always: free, keeps TDM path honest
#endif
  __syncthreads();
}

// ---------------- table prep ----------------
// Forward DFT, transposed [NP][n]: row q<F -> cos(2pi q j/n); F<=q<2F -> -sin; else 0
__global__ void k_dft_table(_Float16* bt, int n, int F, int NP) {
  int idx = blockIdx.x * blockDim.x + threadIdx.x;
  if (idx >= NP * n) return;
  int q = idx / n, j = idx % n;
  float v = 0.f;
  if (q < F)            v =  __cosf(PI2 * (float)(q * j) / (float)n);
  else if (q < 2 * F)   v = -__sinf(PI2 * (float)((q - F) * j) / (float)n);
  bt[idx] = (_Float16)v;
}

// Inverse DFT, [n][KIp]: k=2f -> cf*cos(2pi f j/n)/n ; k=2f+1 -> -2 sin(...)/n (0 for f=0,n/2)
__global__ void k_idft_table(_Float16* it, int n, int F, int KIp) {
  int idx = blockIdx.x * blockDim.x + threadIdx.x;
  if (idx >= n * KIp) return;
  int j = idx / KIp, k = idx % KIp;
  float v = 0.f;
  if (k < 2 * F) {
    int f = k >> 1;
    bool edge = (f == 0) || (f == n / 2);
    float ang = PI2 * (float)(f * j) / (float)n;
    if ((k & 1) == 0) v = (edge ? 1.f : 2.f) * __cosf(ang) / (float)n;
    else              v = edge ? 0.f : (-2.f * __sinf(ang) / (float)n);
  }
  it[idx] = (_Float16)v;
}

// Complexified filter bank, [F][64][32]: rows n<32 -> Re out, n>=32 -> Im out;
// cols k<16 -> Re in (c=k), k>=16 -> Im in (c=k-16)
__global__ void k_fkh(_Float16* fkh, const float* fr, const float* fi, int F) {
  int idx = blockIdx.x * blockDim.x + threadIdx.x;
  if (idx >= F * 64 * 32) return;
  int k = idx & 31, n = (idx >> 5) & 63, f = idx >> 11;
  int o = n & 31, c = k & 15;
  long src = ((long)f * CIN + c) * CO + o;
  float v;
  if (n < 32) v = (k < 16) ? fr[src] : -fi[src];
  else        v = (k < 16) ? fi[src] :  fr[src];
  fkh[idx] = (_Float16)v;
}

// ---------------- framing with reflect pad ----------------
__global__ void k_frames(const float* x, _Float16* A, int T, int n) {
  long idx = (long)blockIdx.x * blockDim.x + threadIdx.x;
  long tot = (long)BATCH * CIN * T * n;
  if (idx >= tot) return;
  int j = (int)(idx % n);
  long row = idx / n;
  int t = (int)(row % T);
  int c = (int)((row / T) % CIN);
  int b = (int)(row / ((long)T * CIN));
  int orig = t * n + j - n / 2;
  if (orig < 0) orig = -orig;
  if (orig >= LEN) orig = 2 * LEN - 2 - orig;
  A[idx] = (_Float16)x[((long)b * CIN + c) * LEN + orig];
}

// ---------------- STFT GEMM: spec = frames x DFT (8 waves share LDS B tile) ----------------
__global__ void k_wmma_stft(const _Float16* __restrict__ A, const _Float16* __restrict__ Bt,
                            float* spre, float* spim, int K, int T, int F) {
  extern __shared__ __align__(16) _Float16 ldsB[];
  int nt = blockIdx.y;
  stage_b_tile(Bt + (long)nt * 16 * K, ldsB, 16 * K);
  int lane = threadIdx.x & 31;
  int mt = blockIdx.x * WPB + (threadIdx.x >> 5);
  const _Float16* arow = A + ((long)mt * 16 + (lane & 15)) * K;
  v8f acc = {};
  for (int kb = 0; kb < K; kb += 32) {
    v16h a = load_a_frag(arow, kb, lane);
    v16h b = load_b_frag(ldsB, K, kb, lane);
    acc = __builtin_amdgcn_wmma_f32_16x16x32_f16(false, a, false, b, (short)0, acc, false, false);
  }
  int q = nt * 16 + (lane & 15);
  int mofs = (lane & 16) ? 8 : 0;
#pragma unroll
  for (int v = 0; v < 8; ++v) {
    int rg = mt * 16 + v + mofs;
    int t = rg % T;
    int c = (rg / T) % CIN;
    int b = rg / (T * CIN);
    float* plane; int f;
    if (q < F)            { plane = spre; f = q; }
    else if (q < 2 * F)   { plane = spim; f = q - F; }
    else continue;
    plane[(((long)b * T + t) * F + f) * CIN + c] = acc[v];
  }
}

// ---------------- band-merge attention (ratio=4) ----------------
__global__ void k_merge(const float* sp0re, const float* sp0im,
                        float* sp1re, float* sp1im,
                        const float* mkr, const float* mki,
                        const float* mbr, const float* mbi) {
  int idx = blockIdx.x * blockDim.x + threadIdx.x;
  const int TOT = BATCH * T1 * F0 * CIN;
  if (idx >= TOT) return;
  int c = idx % CIN;
  int f = (idx / CIN) % F0;
  int t = (idx / (CIN * F0)) % T1;
  int b = idx / (CIN * F0 * T1);
  float pr[RATIO], pi[RATIO];
#pragma unroll
  for (int r = 0; r < RATIO; ++r) {
    long o = (((long)b * T0 + (t * RATIO + r)) * F0 + f) * CIN + c;
    pr[r] = sp0re[o]; pi[r] = sp0im[o];
  }
  float ab[RATIO], mx = -1e30f;
#pragma unroll
  for (int s = 0; s < RATIO; ++s) {
    float are = mbr[s], aim = mbi[s];
#pragma unroll
    for (int r = 0; r < RATIO; ++r) {
      float kr = mkr[r * RATIO + s], ki = mki[r * RATIO + s];
      are += pr[r] * kr - pi[r] * ki;
      aim += pr[r] * ki + pi[r] * kr;
    }
    ab[s] = __fsqrt_rn(are * are + aim * aim);
    mx = fmaxf(mx, ab[s]);
  }
  float se = 0.f, w[RATIO];
#pragma unroll
  for (int s = 0; s < RATIO; ++s) { w[s] = __expf(ab[s] - mx); se += w[s]; }
  float mre = 0.f, mim = 0.f;
#pragma unroll
  for (int r = 0; r < RATIO; ++r) {
    float a = w[r] / se;
    mre += pr[r] * a; mim += pi[r] * a;
  }
  mre *= (float)RATIO; mim *= (float)RATIO;
  long o1 = (((long)b * T1 + t) * F1 + f * RATIO) * CIN + c;
  sp1re[o1] = mre; sp1im[o1] = mim;
}

// ---------------- complex BN statistics per batch element ----------------
__global__ void k_bnstats(const float* re, const float* im,
                          const float* gamma, const float* beta,
                          float* stats, int Nper) {
  __shared__ float sh[3 * 256];
  int b = blockIdx.x;
  const float* pr = re + (long)b * Nper;
  const float* pi = im + (long)b * Nper;
  float a1 = 0.f, a2 = 0.f, a3 = 0.f;
  for (int i = threadIdx.x; i < Nper; i += blockDim.x) {
    float r = pr[i], q = pi[i];
    a1 += r; a2 += q; a3 += r * r + q * q;
  }
  int t = threadIdx.x;
  sh[t] = a1; sh[256 + t] = a2; sh[512 + t] = a3;
  __syncthreads();
  for (int s = 128; s > 0; s >>= 1) {
    if (t < s) { sh[t] += sh[t + s]; sh[256 + t] += sh[256 + t + s]; sh[512 + t] += sh[512 + t + s]; }
    __syncthreads();
  }
  if (t == 0) {
    float n = (float)Nper;
    float mre = sh[0] / n, mim = sh[256] / n;
    float var = sh[512] / n - mre * mre - mim * mim;
    float inv = __frsqrt_rn(var + EPS);
    stats[b * 4 + 0] = mre;
    stats[b * 4 + 1] = mim;
    stats[b * 4 + 2] = inv * gamma[b];
    stats[b * 4 + 3] = beta[b];
  }
}

// ---------------- BN apply -> f16 einsum A matrix [F][M][32] ----------------
__global__ void k_bnapply(const float* re, const float* im, const float* stats,
                          _Float16* Ae, int T, int F) {
  long idx = (long)blockIdx.x * blockDim.x + threadIdx.x;
  long tot = (long)F * BATCH * T * 32;
  if (idx >= tot) return;
  int k = (int)(idx & 31);
  long m = (idx >> 5) % (BATCH * T);
  int f = (int)(idx / (32L * BATCH * T));
  int b = (int)(m / T), t = (int)(m % T);
  const float* st = stats + b * 4;
  long so = (((long)b * T + t) * F + f) * CIN + (k & 15);
  float v;
  if (k < 16) v = (re[so] - st[0]) * st[2] + st[3];
  else        v = (im[so] - st[1]) * st[2];
  Ae[idx] = (_Float16)v;
}

// ---------------- per-frequency channel-mix GEMM: M x 32 x 64 ----------------
__global__ void k_wmma_einsum(const _Float16* __restrict__ Ae, const _Float16* __restrict__ fkh,
                              _Float16* eo, int T, int KIp) {
  extern __shared__ __align__(16) _Float16 ldsB[];
  int f = blockIdx.x, nt = blockIdx.z;
  stage_b_tile(fkh + ((long)f * 64 + nt * 16) * 32, ldsB, 16 * 32);
  int lane = threadIdx.x & 31;
  int mt = blockIdx.y * WPB + (threadIdx.x >> 5);
  long M = (long)BATCH * T;
  const _Float16* arow = Ae + ((long)f * M + mt * 16 + (lane & 15)) * 32;
  v16h a = load_a_frag(arow, 0, lane);
  v16h b = load_b_frag(ldsB, 32, 0, lane);
  v8f acc = {};
  acc = __builtin_amdgcn_wmma_f32_16x16x32_f16(false, a, false, b, (short)0, acc, false, false);
  int n = nt * 16 + (lane & 15);
  int o = n & 31, reim = n >> 5;
  int mofs = (lane & 16) ? 8 : 0;
#pragma unroll
  for (int v = 0; v < 8; ++v) {
    int m = mt * 16 + v + mofs;
    int b_ = m / T, t = m % T;
    eo[(((long)b_ * CO + o) * T + t) * KIp + 2 * f + reim] = (_Float16)acc[v];
  }
}

// ---------------- ISTFT GEMM + crop + bias + relu + transpose ----------------
__global__ void k_wmma_istft(const _Float16* __restrict__ eo, const _Float16* __restrict__ idftT,
                             const float* pbias, float* dout,
                             int T, int Nfft, int KIp, int headOff) {
  extern __shared__ __align__(16) _Float16 ldsB[];
  int nt = blockIdx.y;
  stage_b_tile(idftT + (long)nt * 16 * KIp, ldsB, 16 * KIp);
  int lane = threadIdx.x & 31;
  int mt = blockIdx.x * WPB + (threadIdx.x >> 5);
  int Tp1 = T + 1;
  int rA = mt * 16 + (lane & 15);
  int tA = rA % Tp1;
  int bo = rA / Tp1;
  if (tA == T) tA = 0;                       // wrapped last frame = frame 0
  const _Float16* arow = eo + (((long)(bo / CO) * CO + (bo % CO)) * T + tA) * KIp;
  v8f acc = {};
  for (int kb = 0; kb < KIp; kb += 32) {
    __builtin_prefetch((const void*)(arow + kb + 32));
    v16h a = load_a_frag(arow, kb, lane);
    v16h b = load_b_frag(ldsB, KIp, kb, lane);
    acc = __builtin_amdgcn_wmma_f32_16x16x32_f16(false, a, false, b, (short)0, acc, false, false);
  }
  int j = nt * 16 + (lane & 15);
  int mofs = (lane & 16) ? 8 : 0;
#pragma unroll
  for (int v = 0; v < 8; ++v) {
    int r = mt * 16 + v + mofs;
    int t = r % Tp1;
    int bo2 = r / Tp1;
    int o = bo2 % CO, b_ = bo2 / CO;
    int tau = t * Nfft + j - Nfft / 2;
    if (tau < 0 || tau >= LEN) continue;
    int ch = headOff + o;
    float val = acc[v] + pbias[ch];
    val = val > 0.f ? val : 0.f;
    dout[((long)b_ * (2 * CO) + ch) * LEN + tau] = val;
  }
}

// ---------------- launch ----------------
extern "C" void kernel_launch(void* const* d_in, const int* in_sizes, int n_in,
                              void* d_out, int out_size, void* d_ws, size_t ws_size,
                              hipStream_t stream) {
  const float* x        = (const float*)d_in[0];
  const float* bn_gamma = (const float*)d_in[1];
  const float* bn_beta  = (const float*)d_in[2];
  const float* mk_r     = (const float*)d_in[3];
  const float* mk_i     = (const float*)d_in[4];
  const float* mb_r     = (const float*)d_in[5];
  const float* mb_i     = (const float*)d_in[6];
  const float* fk0_r    = (const float*)d_in[7];
  const float* fk0_i    = (const float*)d_in[8];
  const float* fk1_r    = (const float*)d_in[9];
  const float* fk1_i    = (const float*)d_in[10];
  const float* pbias    = (const float*)d_in[11];
  float* out = (float*)d_out;

  char* w = (char*)d_ws;
  size_t off = 0;
  auto alloc = [&](size_t bytes) -> char* {
    char* p = w + off;
    off = (off + bytes + 255) & ~(size_t)255;
    return p;
  };
  _Float16* dft0T  = (_Float16*)alloc((size_t)NP0 * N0 * 2);
  _Float16* dft1T  = (_Float16*)alloc((size_t)NP1 * N1 * 2);
  _Float16* idft0T = (_Float16*)alloc((size_t)N0 * KI0 * 2);
  _Float16* idft1T = (_Float16*)alloc((size_t)N1 * KI1 * 2);
  _Float16* fkh0   = (_Float16*)alloc((size_t)F0 * 64 * 32 * 2);
  _Float16* fkh1   = (_Float16*)alloc((size_t)F1 * 64 * 32 * 2);
  float*    stats0 = (float*)alloc((size_t)BATCH * 4 * 4);
  float*    stats1 = (float*)alloc((size_t)BATCH * 4 * 4);
  _Float16* A0h    = (_Float16*)alloc((size_t)R0 * N0 * 2);
  _Float16* A1h    = (_Float16*)alloc((size_t)R1 * N1 * 2);
  float*    sp0re  = (float*)alloc((size_t)BATCH * T0 * F0 * CIN * 4);
  float*    sp0im  = (float*)alloc((size_t)BATCH * T0 * F0 * CIN * 4);
  float*    sp1re  = (float*)alloc((size_t)BATCH * T1 * F1 * CIN * 4);
  float*    sp1im  = (float*)alloc((size_t)BATCH * T1 * F1 * CIN * 4);
  _Float16* Ae0    = (_Float16*)alloc((size_t)F0 * EM0 * 32 * 2);
  _Float16* Ae1    = (_Float16*)alloc((size_t)F1 * EM1 * 32 * 2);
  size_t szEo0 = (size_t)BATCH * CO * T0 * KI0 * 2;
  size_t szEo1 = (size_t)BATCH * CO * T1 * KI1 * 2;
  _Float16* eo0 = (_Float16*)alloc(szEo0);
  _Float16* eo1 = (_Float16*)alloc(szEo1);

  auto g1 = [](long n) { return dim3((unsigned)((n + 255) / 256)); };

  // 1) tables
  k_dft_table<<<g1((long)NP0 * N0), 256, 0, stream>>>(dft0T, N0, F0, NP0);
  k_dft_table<<<g1((long)NP1 * N1), 256, 0, stream>>>(dft1T, N1, F1, NP1);
  k_idft_table<<<g1((long)N0 * KI0), 256, 0, stream>>>(idft0T, N0, F0, KI0);
  k_idft_table<<<g1((long)N1 * KI1), 256, 0, stream>>>(idft1T, N1, F1, KI1);
  k_fkh<<<g1((long)F0 * 64 * 32), 256, 0, stream>>>(fkh0, fk0_r, fk0_i, F0);
  k_fkh<<<g1((long)F1 * 64 * 32), 256, 0, stream>>>(fkh1, fk1_r, fk1_i, F1);

  // 2) framing
  k_frames<<<g1((long)R0 * N0), 256, 0, stream>>>(x, A0h, T0, N0);
  k_frames<<<g1((long)R1 * N1), 256, 0, stream>>>(x, A1h, T1, N1);

  // 3) STFT GEMMs (WMMA, 8 waves/block, LDS-staged B)
  k_wmma_stft<<<dim3(R0 / 16 / WPB, NP0 / 16), 256, 16 * N0 * 2, stream>>>(
      A0h, dft0T, sp0re, sp0im, N0, T0, F0);
  k_wmma_stft<<<dim3(R1 / 16 / WPB, NP1 / 16), 256, 16 * N1 * 2, stream>>>(
      A1h, dft1T, sp1re, sp1im, N1, T1, F1);

  // 4) ratio-band merge attention -> writes acc1 bins into sp1 planes
  k_merge<<<g1((long)BATCH * T1 * F0 * CIN), 256, 0, stream>>>(
      sp0re, sp0im, sp1re, sp1im, mk_r, mk_i, mb_r, mb_i);

  // 5) complex BN stats
  k_bnstats<<<BATCH, 256, 0, stream>>>(sp0re, sp0im, bn_gamma, bn_beta, stats0, T0 * F0 * CIN);
  k_bnstats<<<BATCH, 256, 0, stream>>>(sp1re, sp1im, bn_gamma + BATCH, bn_beta + BATCH, stats1, T1 * F1 * CIN);

  // 6) BN apply -> f16 A matrices
  k_bnapply<<<g1((long)F0 * EM0 * 32), 256, 0, stream>>>(sp0re, sp0im, stats0, Ae0, T0, F0);
  k_bnapply<<<g1((long)F1 * EM1 * 32), 256, 0, stream>>>(sp1re, sp1im, stats1, Ae1, T1, F1);

  // 7) zero K-padding of einsum outputs, then per-frequency channel GEMMs (WMMA)
  (void)hipMemsetAsync(eo0, 0, szEo0, stream);
  (void)hipMemsetAsync(eo1, 0, szEo1, stream);
  k_wmma_einsum<<<dim3(F0, EM0 / 16 / WPB, 4), 256, 16 * 32 * 2, stream>>>(Ae0, fkh0, eo0, T0, KI0);
  k_wmma_einsum<<<dim3(F1, EM1 / 16 / WPB, 4), 256, 16 * 32 * 2, stream>>>(Ae1, fkh1, eo1, T1, KI1);

  // 8) ISTFT GEMMs (WMMA) fused with crop/bias/relu/transpose -> d_out
  k_wmma_istft<<<dim3((BATCH * CO * (T0 + 1)) / 16 / WPB, N0 / 16), 256, 16 * KI0 * 2, stream>>>(
      eo0, idft0T, pbias, out, T0, N0, KI0, 0);
  k_wmma_istft<<<dim3((BATCH * CO * (T1 + 1)) / 16 / WPB, N1 / 16), 256, 16 * KI1 * 2, stream>>>(
      eo1, idft1T, pbias, out, T1, N1, KI1, CO);
}
